// Decoder_37804302139817
// MI455X (gfx1250) — compile-verified, matrix operand order checked by reading, not run on hardware
//
#include <hip/hip_runtime.h>
#include <hip/hip_bf16.h>

// ---------------------------------------------------------------------------
// Problem sizes (fixed by the reference)
// ---------------------------------------------------------------------------
enum : int {
    Bn = 128,   // batch
    Sn = 256,   // source length
    Tn = 128,   // target length
    En = 1024,  // embed dim
    Hn = 1024,  // hidden dim
    H2 = 2048,  // 2*H
    H3 = 3072,  // 3*H  (GRU gates)
    GIN = 3072, // E + 2H (GRU input dim)
    PIN = 4096  // 3H + E (pre_output input dim)
};

typedef __attribute__((ext_vector_type(16))) __bf16 v16bf;
typedef __attribute__((ext_vector_type(8)))  float  v8f;
typedef __attribute__((ext_vector_type(4)))  unsigned int su4;
typedef __attribute__((ext_vector_type(8)))  unsigned int su8;
typedef int vsi4 __attribute__((__vector_size__(16)));  // matches builtin V4i

#define AS1 __attribute__((address_space(1)))
#define AS3 __attribute__((address_space(3)))

// Feature detection: async memory->LDS copies (ASYNCcnt path)
#if defined(__has_builtin)
#  if __has_builtin(__builtin_amdgcn_global_load_async_to_lds_b128)
#    define HAVE_ASYNC_LDS 1
#  else
#    define HAVE_ASYNC_LDS 0
#  endif
#  if __has_builtin(__builtin_amdgcn_s_wait_asynccnt)
#    define HAVE_WAIT_ASYNC 1
#  else
#    define HAVE_WAIT_ASYNC 0
#  endif
#else
#  define HAVE_ASYNC_LDS 0
#  define HAVE_WAIT_ASYNC 0
#endif

__device__ __forceinline__ void wait_asynccnt0() {
#if HAVE_WAIT_ASYNC
    __builtin_amdgcn_s_wait_asynccnt(0);
#else
    asm volatile("s_wait_asynccnt 0x0" ::: "memory");
#endif
}

// round-to-nearest-even f32 -> bf16 (stored as ushort)
__device__ __forceinline__ unsigned short f2bf(float x) {
    unsigned int u = __float_as_uint(x);
    u += 0x7FFFu + ((u >> 16) & 1u);
    return (unsigned short)(u >> 16);
}

// ---------------------------------------------------------------------------
// f32 -> bf16 conversion (grid-stride)
// ---------------------------------------------------------------------------
__global__ __launch_bounds__(256) void f32_to_bf16_kernel(
    const float* __restrict__ x, unsigned short* __restrict__ y, long n)
{
    long i = (long)blockIdx.x * blockDim.x + threadIdx.x;
    long stride = (long)gridDim.x * blockDim.x;
    for (; i < n; i += stride) y[i] = f2bf(x[i]);
}

// ---------------------------------------------------------------------------
// Generic bf16 WMMA GEMM:  C[M,N] = act( A[M,K] * Bw[N,K]^T + Cadd + bias )
//   - block tile 128(M) x 64(N), 256 threads = 8 waves (4x2 wave grid)
//   - each wave: 32x32 sub-tile = 2x2 WMMA 16x16 accumulators, K step 32
//   - A tile fed by GLOBAL_LOAD_ASYNC_TO_LDS_B128 (ASYNCcnt)
//   - B tile fed by the Tensor Data Mover (tensor_load_to_lds, TENSORcnt),
//     issued by wave 0 with a D# built per CDNA5 ISA ch.8:
//       2D tile 32(K) x 64(N), data_size=2B, row stride = ldb,
//       pad_enable: 4 DWORDs of pad every 16 DWORDs -> 80B LDS row pitch
// Requirements: M%128==0, N%64==0, K%32==0 (all shapes here satisfy this)
// ---------------------------------------------------------------------------
__global__ __launch_bounds__(256) void gemm_bf16_wmma_kernel(
    const unsigned short* __restrict__ A,  int lda,
    const unsigned short* __restrict__ Bw, int ldb,
    float* __restrict__ C,  int ldc,
    const float* __restrict__ Cadd, int ldca,
    const float* __restrict__ bias, int act,
    unsigned short* __restrict__ Cbf, int ldcbf,
    int K)
{
    // padded LDS tiles: row pitch 40 bf16 (80 B) to stagger banks
    __shared__ unsigned short sA[128 * 40];
    __shared__ unsigned short sB[64 * 40];

    const int tid  = threadIdx.x;
    const int lane = tid & 31;
    const int wave = tid >> 5;
    const int wm   = wave & 3;   // wave row: 0..3 -> 32-row strip
    const int wn   = wave >> 2;  // wave col: 0..1 -> 32-col strip
    const int m0   = blockIdx.y * 128;
    const int n0   = blockIdx.x * 64;
    const int fr   = lane & 15;          // position within 16 (M or N index)
    const int half = (lane >> 4) & 1;    // K-half select per ISA layout

    const unsigned int sB_lds = (unsigned int)(size_t)(AS3 unsigned short*)sB;

    v8f acc00 = {0,0,0,0,0,0,0,0};
    v8f acc01 = {0,0,0,0,0,0,0,0};
    v8f acc10 = {0,0,0,0,0,0,0,0};
    v8f acc11 = {0,0,0,0,0,0,0,0};

    union Frag { uint4 u[2]; v16bf v; };
    union G1 { unsigned int w[8]; su8 v; };

    for (int kk = 0; kk < K; kk += 32) {
        // prefetch next A K-tile rows into L2 (global_prefetch_b8)
        if (kk + 32 < K) {
            __builtin_prefetch(A + (size_t)(m0 + (tid >> 1)) * lda + kk + 32, 0, 1);
        }

        // ---- B tile via Tensor Data Mover (one descriptor, wave 0 issues) ----
        if (tid < 32) {
            unsigned long long ga =
                (unsigned long long)(const void*)(Bw + (size_t)n0 * ldb + kk);
            su4 g0;
            g0.x = 1u;                                   // count=1, user D#
            g0.y = sB_lds;                               // lds_addr (bytes)
            g0.z = (unsigned int)(ga & 0xFFFFFFFFu);     // global_addr[31:0]
            g0.w = (unsigned int)((ga >> 32) & 0x1FFFFFFu) | (2u << 30); // type=2
            G1 g1;
            // data_size=2B (code 1) | pad_enable | pad_interval=16DW (code 3)
            // | pad_amount=4DW (code 3)
            g1.w[0] = (1u << 16) | (1u << 20) | (3u << 22) | (3u << 25);
            g1.w[1] = (32u << 16);         // tensor_dim0 = 32 (bits 79:48 lo)
            g1.w[2] = (64u << 16);         // tensor_dim1 = 64 (bits 111:80 lo)
            g1.w[3] = (32u << 16);         // tile_dim0 = 32 (bits 127:112)
            g1.w[4] = 64u;                 // tile_dim1 = 64 (bits 143:128)
            g1.w[5] = (unsigned int)ldb;   // tensor_dim0_stride[31:0]
            g1.w[6] = 0u;
            g1.w[7] = 0u;
            asm volatile("tensor_load_to_lds %0, %1" :: "s"(g0), "s"(g1.v)
                         : "memory");
        }

        // ---- A tile: 128 rows x 32 bf16 = 512 x 16B chunks ----
#if HAVE_ASYNC_LDS
        #pragma unroll
        for (int it = 0; it < 2; ++it) {
            int c  = tid + it * 256;     // 0..511
            int r  = c >> 2;
            int cc = c & 3;
            __builtin_amdgcn_global_load_async_to_lds_b128(
                (AS1 vsi4*)(A + (size_t)(m0 + r) * lda + kk + cc * 8),
                (AS3 vsi4*)(&sA[r * 40 + cc * 8]), 0, 0);
        }
        wait_asynccnt0();
#else
        #pragma unroll
        for (int it = 0; it < 2; ++it) {
            int c  = tid + it * 256;
            int r  = c >> 2;
            int cc = c & 3;
            uint4 val = *(const uint4*)(A + (size_t)(m0 + r) * lda + kk + cc * 8);
            *(uint4*)(&sA[r * 40 + cc * 8]) = val;
        }
#endif
        if (tid < 32) {
            __builtin_amdgcn_s_wait_tensorcnt(0);  // B tile landed in LDS
        }
        __syncthreads();

        // ---- build fragments (ISA 7.12.2 16-bit A 16x32 layout) ----
        Frag fa0, fa1, fb0, fb1;
        {
            int ra = wm * 32 + fr;
            fa0.u[0] = *(const uint4*)(&sA[ra * 40 + half * 8]);
            fa0.u[1] = *(const uint4*)(&sA[ra * 40 + 16 + half * 8]);
            fa1.u[0] = *(const uint4*)(&sA[(ra + 16) * 40 + half * 8]);
            fa1.u[1] = *(const uint4*)(&sA[(ra + 16) * 40 + 16 + half * 8]);
            int rb = wn * 32 + fr;
            fb0.u[0] = *(const uint4*)(&sB[rb * 40 + half * 8]);
            fb0.u[1] = *(const uint4*)(&sB[rb * 40 + 16 + half * 8]);
            fb1.u[0] = *(const uint4*)(&sB[(rb + 16) * 40 + half * 8]);
            fb1.u[1] = *(const uint4*)(&sB[(rb + 16) * 40 + 16 + half * 8]);
        }
        acc00 = __builtin_amdgcn_wmma_f32_16x16x32_bf16(false, fa0.v, false, fb0.v, (short)0, acc00, false, false);
        acc01 = __builtin_amdgcn_wmma_f32_16x16x32_bf16(false, fa0.v, false, fb1.v, (short)0, acc01, false, false);
        acc10 = __builtin_amdgcn_wmma_f32_16x16x32_bf16(false, fa1.v, false, fb0.v, (short)0, acc10, false, false);
        acc11 = __builtin_amdgcn_wmma_f32_16x16x32_bf16(false, fa1.v, false, fb1.v, (short)0, acc11, false, false);
        __syncthreads();
    }

    // ---- epilogue: C/D 16x16 f32 layout -> global ----
    v8f accs[2][2];
    accs[0][0] = acc00; accs[0][1] = acc01;
    accs[1][0] = acc10; accs[1][1] = acc11;

    #pragma unroll
    for (int mi = 0; mi < 2; ++mi) {
        #pragma unroll
        for (int ni = 0; ni < 2; ++ni) {
            int baseRow = m0 + wm * 32 + mi * 16 + ((lane & 16) ? 8 : 0);
            int col     = n0 + wn * 32 + ni * 16 + fr;
            #pragma unroll
            for (int v = 0; v < 8; ++v) {
                int row = baseRow + v;
                float val = accs[mi][ni][v];
                if (Cadd) val += Cadd[(size_t)row * ldca + col];
                if (bias) val += bias[col];
                if (act)  val = tanhf(val);
                C[(size_t)row * ldc + col] = val;
                if (Cbf) Cbf[(size_t)row * ldcbf + col] = f2bf(val);
            }
        }
    }
}

// ---------------------------------------------------------------------------
// Bahdanau attention for one timestep. One block per batch row b.
// scores_s = sum_h v[h]*tanh(q[b,h] + proj_key[b,s,h]); masked softmax;
// context[b,:] = sum_s alpha_s * enc[b,s,:]
// ---------------------------------------------------------------------------
__global__ __launch_bounds__(256) void attn_kernel(
    const float* __restrict__ q, const float* __restrict__ proj_key,
    const float* __restrict__ enc, const int* __restrict__ mask,
    const float* __restrict__ ve,
    float* __restrict__ ctx, unsigned short* __restrict__ ctx_bf)
{
    __shared__ float sq[Hn];
    __shared__ float sv[Hn];
    __shared__ float sc[Sn];
    __shared__ float red[256];

    const int b    = blockIdx.x;
    const int tid  = threadIdx.x;
    const int lane = tid & 31;
    const int wave = tid >> 5;

    for (int i = tid; i < Hn; i += 256) { sq[i] = q[b * Hn + i]; sv[i] = ve[i]; }
    __syncthreads();

    // scores (8 waves, 32 source positions each)
    for (int s = wave; s < Sn; s += 8) {
        const float* pk = proj_key + ((size_t)b * Sn + s) * Hn;
        float acc = 0.f;
        for (int h = lane; h < Hn; h += 32)
            acc += sv[h] * tanhf(sq[h] + pk[h]);
        #pragma unroll
        for (int off = 16; off > 0; off >>= 1) acc += __shfl_xor(acc, off, 32);
        if (lane == 0)
            sc[s] = (mask[b * Sn + s] != 0) ? acc : -3.402823466e38f;
    }
    __syncthreads();

    // softmax over S=256 (one score per thread)
    float v = sc[tid];
    red[tid] = v; __syncthreads();
    for (int st = 128; st > 0; st >>= 1) {
        if (tid < st) red[tid] = fmaxf(red[tid], red[tid + st]);
        __syncthreads();
    }
    float m = red[0];
    __syncthreads();
    float p = __expf(v - m);
    red[tid] = p; __syncthreads();
    for (int st = 128; st > 0; st >>= 1) {
        if (tid < st) red[tid] += red[tid + st];
        __syncthreads();
    }
    float inv = 1.0f / red[0];
    __syncthreads();
    sc[tid] = p * inv;
    __syncthreads();

    // context: each thread owns 8 consecutive channels of 2H=2048
    const int k0 = tid * 8;
    float a8[8] = {0,0,0,0,0,0,0,0};
    for (int s = 0; s < Sn; ++s) {
        float al = sc[s];
        const float4* e = (const float4*)(enc + ((size_t)b * Sn + s) * H2 + k0);
        float4 e0 = e[0], e1 = e[1];
        a8[0] += al * e0.x; a8[1] += al * e0.y; a8[2] += al * e0.z; a8[3] += al * e0.w;
        a8[4] += al * e1.x; a8[5] += al * e1.y; a8[6] += al * e1.z; a8[7] += al * e1.w;
    }
    #pragma unroll
    for (int i = 0; i < 8; ++i) {
        float vv = a8[i];
        ctx[(size_t)b * H2 + k0 + i]    = vv;
        ctx_bf[(size_t)b * H2 + k0 + i] = f2bf(vv);
    }
}

// ---------------------------------------------------------------------------
// GRU gates (pointwise): r,z,n -> h_new; writes h, h_bf, decoder_states[:,t,:]
// ---------------------------------------------------------------------------
__global__ __launch_bounds__(256) void gru_gate_kernel(
    const float* __restrict__ gi_x,  // [B*T, 3H], row (b*T + t)
    int t,
    const float* __restrict__ gic,   // [B, 3H]
    const float* __restrict__ gh,    // [B, 3H]
    const float* __restrict__ b_ih, const float* __restrict__ b_hh,
    float* __restrict__ h, unsigned short* __restrict__ h_bf,
    float* __restrict__ out_states)  // [B, T, H]
{
    int gidx = blockIdx.x * blockDim.x + threadIdx.x; // 0 .. B*H-1
    int b = gidx / Hn;
    int j = gidx - b * Hn;

    size_t rx = ((size_t)b * Tn + t) * (size_t)H3;
    size_t rb = (size_t)b * H3;

    float ir = gi_x[rx + j]          + gic[rb + j]          + b_ih[j];
    float iz = gi_x[rx + Hn + j]     + gic[rb + Hn + j]     + b_ih[Hn + j];
    float in = gi_x[rx + 2*Hn + j]   + gic[rb + 2*Hn + j]   + b_ih[2*Hn + j];
    float hr = gh[rb + j]          + b_hh[j];
    float hz = gh[rb + Hn + j]     + b_hh[Hn + j];
    float hn = gh[rb + 2*Hn + j]   + b_hh[2*Hn + j];

    float r = 1.0f / (1.0f + __expf(-(ir + hr)));
    float z = 1.0f / (1.0f + __expf(-(iz + hz)));
    float n = tanhf(in + r * hn);
    float hprev = h[(size_t)b * Hn + j];
    float hnew  = (1.0f - z) * n + z * hprev;

    h[(size_t)b * Hn + j]   = hnew;
    h_bf[(size_t)b * Hn + j] = f2bf(hnew);
    out_states[((size_t)b * Tn + t) * Hn + j] = hnew;
}

// ---------------------------------------------------------------------------
// Host orchestration
// ---------------------------------------------------------------------------
static inline char* bump(char*& p, size_t bytes) {
    char* r = p;
    p += (bytes + 255) & ~(size_t)255;
    return r;
}

extern "C" void kernel_launch(void* const* d_in, const int* in_sizes, int n_in,
                              void* d_out, int out_size, void* d_ws, size_t ws_size,
                              hipStream_t stream) {
    (void)in_sizes; (void)n_in; (void)out_size; (void)ws_size;

    const float* trg   = (const float*)d_in[0];   // [B,T,E]
    const float* enc   = (const float*)d_in[1];   // [B,S,2H]
    const float* encf  = (const float*)d_in[2];   // [1,B,2H]
    const int*   mask  = (const int*)d_in[3];     // [B,S]
    const float* Wk    = (const float*)d_in[4];   // [H,2H]
    const float* Wq    = (const float*)d_in[5];   // [H,H]
    const float* ve    = (const float*)d_in[6];   // [H]
    const float* Wih   = (const float*)d_in[7];   // [3H, E+2H]
    const float* Whh   = (const float*)d_in[8];   // [3H, H]
    const float* bih   = (const float*)d_in[9];   // [3H]
    const float* bhh   = (const float*)d_in[10];  // [3H]
    const float* Wbr   = (const float*)d_in[11];  // [H,2H]
    const float* bbr   = (const float*)d_in[12];  // [H]
    const float* Wpre  = (const float*)d_in[13];  // [E, 3H+E]

    float* out        = (float*)d_out;
    float* out_states = out;                              // [B,T,H]
    float* out_hT     = out + (size_t)Bn * Tn * Hn;       // [1,B,H]
    float* out_pre    = out_hT + (size_t)Bn * Hn;         // [B,T,E]

    // ---- workspace layout ----
    char* p = (char*)d_ws;
    unsigned short* wq_bf   = (unsigned short*)bump(p, (size_t)Hn * Hn * 2);
    unsigned short* wk_bf   = (unsigned short*)bump(p, (size_t)Hn * H2 * 2);
    unsigned short* wih_bf  = (unsigned short*)bump(p, (size_t)H3 * GIN * 2);
    unsigned short* whh_bf  = (unsigned short*)bump(p, (size_t)H3 * Hn * 2);
    unsigned short* wpre_bf = (unsigned short*)bump(p, (size_t)En * PIN * 2);
    unsigned short* wbr_bf  = (unsigned short*)bump(p, (size_t)Hn * H2 * 2);
    unsigned short* enc_bf  = (unsigned short*)bump(p, (size_t)Bn * Sn * H2 * 2);
    unsigned short* encf_bf = (unsigned short*)bump(p, (size_t)Bn * H2 * 2);
    unsigned short* trg_bf  = (unsigned short*)bump(p, (size_t)Bn * Tn * En * 2);
    float* proj_key = (float*)bump(p, (size_t)Bn * Sn * Hn * 4);
    float* gi_x     = (float*)bump(p, (size_t)Bn * Tn * H3 * 4);
    float* pre_x    = (float*)bump(p, (size_t)Bn * Tn * En * 4);
    float* hbuf     = (float*)bump(p, (size_t)Bn * Hn * 4);
    unsigned short* h_bf   = (unsigned short*)bump(p, (size_t)Bn * Hn * 2);
    float* qbuf     = (float*)bump(p, (size_t)Bn * Hn * 4);
    float* ctx      = (float*)bump(p, (size_t)Bn * H2 * 4);
    unsigned short* ctx_bf = (unsigned short*)bump(p, (size_t)Bn * H2 * 2);
    float* gh_buf   = (float*)bump(p, (size_t)Bn * H3 * 4);
    float* gic_buf  = (float*)bump(p, (size_t)Bn * H3 * 4);

    // ---- bf16 conversions ----
    auto conv = [&](const float* src, unsigned short* dst, long n) {
        f32_to_bf16_kernel<<<dim3(2048), dim3(256), 0, stream>>>(src, dst, n);
    };
    conv(Wq,   wq_bf,   (long)Hn * Hn);
    conv(Wk,   wk_bf,   (long)Hn * H2);
    conv(Wih,  wih_bf,  (long)H3 * GIN);
    conv(Whh,  whh_bf,  (long)H3 * Hn);
    conv(Wpre, wpre_bf, (long)En * PIN);
    conv(Wbr,  wbr_bf,  (long)Hn * H2);
    conv(enc,  enc_bf,  (long)Bn * Sn * H2);
    conv(encf, encf_bf, (long)Bn * H2);
    conv(trg,  trg_bf,  (long)Bn * Tn * En);

    // ---- generic GEMM launcher ----
    auto gemm = [&](const unsigned short* A, int lda,
                    const unsigned short* Bw, int ldb,
                    float* C, int ldc,
                    const float* Cadd, int ldca,
                    const float* bias, int act,
                    unsigned short* Cbf, int ldcbf,
                    int M, int N, int K) {
        dim3 grid(N / 64, M / 128);
        gemm_bf16_wmma_kernel<<<grid, dim3(256), 0, stream>>>(
            A, lda, Bw, ldb, C, ldc, Cadd, ldca, bias, act, Cbf, ldcbf, K);
    };

    // h0 = tanh(encoder_final @ Wbr^T + bbr)
    gemm(encf_bf, H2, wbr_bf, H2, hbuf, Hn, nullptr, 0, bbr, 1, h_bf, Hn,
         Bn, Hn, H2);
    // proj_key = encoder_hidden @ Wk^T   [B*S, H]
    gemm(enc_bf, H2, wk_bf, H2, proj_key, Hn, nullptr, 0, nullptr, 0, nullptr, 0,
         Bn * Sn, Hn, H2);
    // gi_x = trg_embed @ Wih[:, :E]^T   [B*T, 3H]
    gemm(trg_bf, En, wih_bf, GIN, gi_x, H3, nullptr, 0, nullptr, 0, nullptr, 0,
         Bn * Tn, H3, En);
    // pre_x = trg_embed @ Wpre[:, :E]^T [B*T, E]
    gemm(trg_bf, En, wpre_bf, PIN, pre_x, En, nullptr, 0, nullptr, 0, nullptr, 0,
         Bn * Tn, En, En);

    // ---- sequential scan over T ----
    for (int t = 0; t < Tn; ++t) {
        // q = h @ Wq^T
        gemm(h_bf, Hn, wq_bf, Hn, qbuf, Hn, nullptr, 0, nullptr, 0, nullptr, 0,
             Bn, Hn, Hn);
        // attention -> context
        attn_kernel<<<dim3(Bn), dim3(256), 0, stream>>>(
            qbuf, proj_key, enc, mask, ve, ctx, ctx_bf);
        // gh = h @ Whh^T
        gemm(h_bf, Hn, whh_bf, Hn, gh_buf, H3, nullptr, 0, nullptr, 0, nullptr, 0,
             Bn, H3, Hn);
        // gi_ctx = context @ Wih[:, E:]^T
        gemm(ctx_bf, H2, wih_bf + En, GIN, gic_buf, H3, nullptr, 0, nullptr, 0,
             nullptr, 0, Bn, H3, H2);
        // gates -> h_new (writes decoder_states[:, t, :])
        gru_gate_kernel<<<dim3((Bn * Hn) / 256), dim3(256), 0, stream>>>(
            gi_x, t, gic_buf, gh_buf, bih, bhh, hbuf, h_bf, out_states);
        // pre = pre_x[:,t,:] + h_new @ Wpre[:,E:E+H]^T
        gemm(h_bf, Hn, wpre_bf + En, PIN, out_pre + (size_t)t * En, Tn * En,
             pre_x + (size_t)t * En, Tn * En, nullptr, 0, nullptr, 0,
             Bn, En, Hn);
        // pre += context @ Wpre[:,E+H:]^T
        gemm(ctx_bf, H2, wpre_bf + (En + Hn), PIN, out_pre + (size_t)t * En, Tn * En,
             out_pre + (size_t)t * En, Tn * En, nullptr, 0, nullptr, 0,
             Bn, En, H2);
    }

    // hT = final hidden state
    (void)hipMemcpyAsync(out_hT, hbuf, (size_t)Bn * Hn * sizeof(float),
                         hipMemcpyDeviceToDevice, stream);
}
